// ApproachNet_41729902248456
// MI455X (gfx1250) — matrix operand order repaired and checked by reading, not compiled, not run on hardware
//
#include <hip/hip_runtime.h>
#include <hip/hip_bf16.h>
#include <math.h>

typedef __attribute__((ext_vector_type(16))) _Float16 v16h;
typedef __attribute__((ext_vector_type(8)))  _Float16 v8h;
typedef __attribute__((ext_vector_type(8)))  float    v8f;

#define BB   8
#define PP   8192
#define SS1  1639
#define SS2  410
#define MAXNB 64
#define GFD  1024

// ---------------------------------------------------------------------------
// WMMA helper: D = A(16x32 f16) x B(32x16 f16) + C(16x16 f32)
// ---------------------------------------------------------------------------
__device__ __forceinline__ v8f wmma_f16(v16h a, v16h b, v8f c) {
  return __builtin_amdgcn_wmma_f32_16x16x32_f16(false, a, false, b, (short)0, c,
                                                false, false);
}

__device__ __forceinline__ v16h cat8(v8h lo, v8h hi) {
  return __builtin_shufflevector(lo, hi, 0, 1, 2, 3, 4, 5, 6, 7,
                                         8, 9, 10, 11, 12, 13, 14, 15);
}

// ---------------------------------------------------------------------------
// Weight swizzle: W[K,N] f32 -> Wsw f16 in this file's B-fragment order (k contiguous per lane half, as A):
//   Wsw[((kc*NT + nt)*32 + lane)*16 + hi] = W[kc*32 + (lane>>4)*16 + hi][nt*16 + (lane&15)]
// K zero-padded to a multiple of 32. Run once per layer per launch.
// ---------------------------------------------------------------------------
__global__ void swz_weight_kernel(const float* __restrict__ W, _Float16* __restrict__ Wsw,
                                  int K, int N) {
  const int NT = N >> 4;
  const int KP = ((K + 31) >> 5) << 5;
  const long total = (long)KP * N;
  for (long e = blockIdx.x * (long)blockDim.x + threadIdx.x; e < total;
       e += (long)gridDim.x * blockDim.x) {
    const int hi = (int)(e & 15);
    const int l  = (int)((e >> 4) & 31);
    const long rest = e >> 9;            // kc*NT + nt
    const int nt = (int)(rest % NT);
    const int kc = (int)(rest / NT);
    const int k = kc * 32 + (l >> 4) * 16 + hi;
    const int n = nt * 16 + (l & 15);
    Wsw[e] = (_Float16)((k < K) ? W[(size_t)k * N + n] : 0.0f);
  }
}

// ---------------------------------------------------------------------------
// Farthest point sampling: one block per cloud, dists cached in LDS.
// ---------------------------------------------------------------------------
__global__ void __launch_bounds__(256)
fps_kernel(const float* __restrict__ pos, int N, int n,
           int* __restrict__ idx, float* __restrict__ spos) {
  const int b = blockIdx.x, t = threadIdx.x;
  const float* pb = pos + (size_t)b * N * 3;
  __shared__ float dists[PP];
  __shared__ float rv[256];
  __shared__ int   ri[256];
  __shared__ float lp[3];
  for (int j = t; j < N; j += 256) dists[j] = 3.4e38f;
  if (t == 0) {
    idx[(size_t)b * n] = 0;
    lp[0] = pb[0]; lp[1] = pb[1]; lp[2] = pb[2];
    spos[((size_t)b * n) * 3 + 0] = pb[0];
    spos[((size_t)b * n) * 3 + 1] = pb[1];
    spos[((size_t)b * n) * 3 + 2] = pb[2];
  }
  for (int i = 1; i < n; ++i) {
    __syncthreads();
    const float bx = lp[0], by = lp[1], bz = lp[2];
    float bestv = -1.0f; int besti = N - 1;
    for (int j = t; j < N; j += 256) {
      const float dx = pb[3*j+0]-bx, dy = pb[3*j+1]-by, dz = pb[3*j+2]-bz;
      const float d = dx*dx + dy*dy + dz*dz;
      const float m = fminf(dists[j], d);
      dists[j] = m;
      if (m > bestv) { bestv = m; besti = j; }
    }
    rv[t] = bestv; ri[t] = besti;
    __syncthreads();
    for (int s = 128; s > 0; s >>= 1) {
      if (t < s) {
        if (rv[t+s] > rv[t] || (rv[t+s] == rv[t] && ri[t+s] < ri[t])) {
          rv[t] = rv[t+s]; ri[t] = ri[t+s];
        }
      }
      __syncthreads();
    }
    if (t == 0) {
      const int j = ri[0];
      idx[(size_t)b*n + i] = j;
      lp[0] = pb[3*j]; lp[1] = pb[3*j+1]; lp[2] = pb[3*j+2];
      spos[((size_t)b*n + i)*3 + 0] = lp[0];
      spos[((size_t)b*n + i)*3 + 1] = lp[1];
      spos[((size_t)b*n + i)*3 + 2] = lp[2];
    }
  }
}

// ---------------------------------------------------------------------------
// Radius neighbors: <=64 nearest within r (binary-searched distance threshold,
// deterministic single-thread emission; order irrelevant for max-pool).
// ---------------------------------------------------------------------------
__global__ void __launch_bounds__(256)
neighbor_kernel(const float* __restrict__ spos, const float* __restrict__ pos,
                int N, int nc, float r2,
                int* __restrict__ nbr, int* __restrict__ nvalid) {
  const int c = blockIdx.x, t = threadIdx.x;
  const int b = c / nc;
  const float* pb = pos + (size_t)b * N * 3;
  const float cx = spos[3*c], cy = spos[3*c+1], cz = spos[3*c+2];
  __shared__ float d2s[PP];
  __shared__ int red[256];
  for (int j = t; j < N; j += 256) {
    const float dx = pb[3*j]-cx, dy = pb[3*j+1]-cy, dz = pb[3*j+2]-cz;
    d2s[j] = dx*dx + dy*dy + dz*dz;
  }
  __syncthreads();
  int cl = 0;
  for (int j = t; j < N; j += 256) if (d2s[j] <= r2) cl++;
  red[t] = cl; __syncthreads();
  for (int s = 128; s > 0; s >>= 1) { if (t < s) red[t] += red[t+s]; __syncthreads(); }
  const int total = red[0];
  __syncthreads();
  float thr = r2;
  if (total > MAXNB) {
    float lo = 0.0f, hi = r2;
    for (int it = 0; it < 24; ++it) {
      const float mid = 0.5f * (lo + hi);
      int cm = 0;
      for (int j = t; j < N; j += 256) if (d2s[j] <= mid) cm++;
      red[t] = cm; __syncthreads();
      for (int s = 128; s > 0; s >>= 1) { if (t < s) red[t] += red[t+s]; __syncthreads(); }
      const int cnt = red[0];
      __syncthreads();
      if (cnt <= MAXNB) lo = mid; else hi = mid;
    }
    thr = lo;
  }
  if (t == 0) {
    int s = 0;
    for (int j = 0; j < N && s < MAXNB; ++j)
      if (d2s[j] <= thr) nbr[(size_t)c*MAXNB + (s++)] = j;
    if (total > MAXNB) {
      for (int j = 0; j < N && s < MAXNB; ++j)
        if (d2s[j] > thr && d2s[j] <= r2) nbr[(size_t)c*MAXNB + (s++)] = j;
    }
    nvalid[c] = s;
  }
}

// ---------------------------------------------------------------------------
// Wave-local WMMA layer: Out rows [wv*16, wv*16+16) = act(Ain x W + b).
// A fragments: two aligned 16B LDS loads; B fragments: contiguous 32B loads
// from pre-swizzled global weights (L2-resident). No barriers needed: all
// reads/writes are confined to the wave's own 16-row stripe.
// ---------------------------------------------------------------------------
template<int KPAD, int COUT>
__device__ __forceinline__ void
block_layer_sw(const _Float16* Ain, const _Float16* __restrict__ Wsw,
               const float* __restrict__ bias, bool relu, _Float16* Out) {
  const int lane = threadIdx.x & 31;
  const int wv   = threadIdx.x >> 5;
  const int g    = lane >> 4;
  const int ln   = lane & 15;
  const int m0   = wv * 16;
  constexpr int NT = COUT / 16;
  for (int nt = 0; nt < NT; ++nt) {
    v8f acc = {};
    for (int kc = 0; kc < KPAD / 32; ++kc) {
      const _Float16* ap = Ain + (m0 + ln) * KPAD + kc * 32 + g * 16;
      const v16h af = cat8(*(const v8h*)ap, *(const v8h*)(ap + 8));
      const v16h bf = *(const v16h*)(Wsw + ((size_t)(kc * NT + nt) * 32 + lane) * 16);
      acc = wmma_f16(af, bf, acc);
    }
#pragma unroll
    for (int r = 0; r < 8; ++r) {
      const int m = m0 + g*8 + r;
      const int n = nt*16 + ln;
      float v = acc[r] + bias[n];
      if (relu) v = fmaxf(v, 0.0f);
      Out[m * COUT + n] = (_Float16)v;
    }
  }
}

// ---------------------------------------------------------------------------
// Fused set-abstraction MLP: one block (4 waves) per center. Rows are wave-
// partitioned through gather + 3 WMMA layers (barrier-free); single barrier
// before the cross-wave masked max-pool.
// ---------------------------------------------------------------------------
template<int CFEAT, int KP, int C1, int C2, int C3>
__global__ void __launch_bounds__(128)
sa_mlp_kernel(const float* __restrict__ xsrc, const float* __restrict__ psrc,
              int N, int nc, const float* __restrict__ spos,
              const int* __restrict__ nbr, const int* __restrict__ nvalid,
              const _Float16* __restrict__ Wsw1, const float* __restrict__ b1,
              const _Float16* __restrict__ Wsw2, const float* __restrict__ b2,
              const _Float16* __restrict__ Wsw3, const float* __restrict__ b3,
              float* __restrict__ xout) {
  constexpr int B0 = (KP > C2) ? KP : C2;
  constexpr int B1 = (C1 > C3) ? C1 : C3;
  __shared__ alignas(32) _Float16 Buf0[64 * B0];
  __shared__ alignas(32) _Float16 Buf1[64 * B1];
  const int c = blockIdx.x;
  const int b = c / nc;
  const int nv = nvalid[c];
  const int lane = threadIdx.x & 31, wv = threadIdx.x >> 5;
  const int m0 = wv * 16;
  // wave-local gather of rows [m0, m0+16): concat(x[nbr], p[nbr]-pq), zero pad
  for (int e = lane; e < 16 * KP; e += 32) {
    const int row = m0 + e / KP, k = e % KP;
    float v = 0.0f;
    if (row < nv) {
      const int s = nbr[(size_t)c * MAXNB + row];
      if (k < CFEAT)
        v = xsrc[((size_t)b * N + s) * CFEAT + k];
      else if (k < CFEAT + 3)
        v = psrc[((size_t)b * N + s) * 3 + (k - CFEAT)] - spos[(size_t)c*3 + (k - CFEAT)];
    }
    Buf0[row * KP + k] = (_Float16)v;
  }
  block_layer_sw<KP, C1>(Buf0, Wsw1, b1, true,  Buf1);
  block_layer_sw<C1, C2>(Buf1, Wsw2, b2, true,  Buf0);
  block_layer_sw<C2, C3>(Buf0, Wsw3, b3, false, Buf1);
  __syncthreads();
  for (int ch = threadIdx.x; ch < C3; ch += 128) {
    float m = -3.4e38f;
    for (int r = 0; r < nv; ++r) m = fmaxf(m, (float)Buf1[r * C3 + ch]);
    xout[(size_t)c * C3 + ch] = m;
  }
}

// ---------------------------------------------------------------------------
// Barrier-free tiled WMMA GEMM: C[M,N] = act(A[M,K] x W[K,N] + b).
// 64x64 block tile, 4 waves each owning a 16-row stripe. Each wave stages its
// own A rows (f32->f16) into a private LDS slice (same-wave DS is in-order),
// B fragments stream from pre-swizzled global weights. N must be a multiple
// of 64 at all call sites; M and K edges are guarded/zero-padded.
// ---------------------------------------------------------------------------
#define ASTR 40
__global__ void __launch_bounds__(128)
gemm_sw_kernel(const float* __restrict__ A, const _Float16* __restrict__ Wsw,
               const float* __restrict__ bias, float* __restrict__ C,
               int M, int N, int K, int relu) {
  __shared__ alignas(32) _Float16 As[64 * ASTR];
  const int lane = threadIdx.x & 31, wv = threadIdx.x >> 5;
  const int g = lane >> 4, ln = lane & 15;
  const int m0 = wv * 16;
  const int mr0 = blockIdx.y * 64, nc0 = blockIdx.x * 64;
  const int NT = N >> 4;
  const int nt0 = nc0 >> 4;
  const int KP = ((K + 31) >> 5) << 5;
  v8f acc0 = {}, acc1 = {}, acc2 = {}, acc3 = {};
  const int srow = m0 + (lane >> 1);      // wave-local staging row
  const int sc0  = (lane & 1) * 16;
  const int gr   = mr0 + srow;
  for (int kb = 0; kb < KP; kb += 32) {
    if (gr < M && kb + 64 < K) __builtin_prefetch(&A[(size_t)gr * K + kb + 64], 0, 0);
#pragma unroll
    for (int u = 0; u < 16; ++u) {
      const int k = kb + sc0 + u;
      const float v = (gr < M && k < K) ? A[(size_t)gr * K + k] : 0.0f;
      As[srow * ASTR + sc0 + u] = (_Float16)v;
    }
    const _Float16* ap = As + (m0 + ln) * ASTR + g * 16;
    const v16h af = cat8(*(const v8h*)ap, *(const v8h*)(ap + 8));
    const _Float16* wb = Wsw + ((size_t)((kb >> 5) * NT + nt0) * 32 + lane) * 16;
    v16h bf;
    bf = *(const v16h*)(wb +    0); acc0 = wmma_f16(af, bf, acc0);
    bf = *(const v16h*)(wb +  512); acc1 = wmma_f16(af, bf, acc1);
    bf = *(const v16h*)(wb + 1024); acc2 = wmma_f16(af, bf, acc2);
    bf = *(const v16h*)(wb + 1536); acc3 = wmma_f16(af, bf, acc3);
  }
#pragma unroll
  for (int r = 0; r < 8; ++r) {
    const int m = mr0 + m0 + g*8 + r;
    if (m < M) {
      const int n0 = nc0 + ln;
      { float v = acc0[r] + bias[n0];      if (relu) v = fmaxf(v, 0.f); C[(size_t)m*N + n0]      = v; }
      { float v = acc1[r] + bias[n0 + 16]; if (relu) v = fmaxf(v, 0.f); C[(size_t)m*N + n0 + 16] = v; }
      { float v = acc2[r] + bias[n0 + 32]; if (relu) v = fmaxf(v, 0.f); C[(size_t)m*N + n0 + 32] = v; }
      { float v = acc3[r] + bias[n0 + 48]; if (relu) v = fmaxf(v, 0.f); C[(size_t)m*N + n0 + 48] = v; }
    }
  }
}

// ---------------------------------------------------------------------------
// Small VALU linear for tiny N (head final, appenc, grasppred final).
// ---------------------------------------------------------------------------
__global__ void linear_small_kernel(const float* __restrict__ A, const float* __restrict__ W,
                                    const float* __restrict__ bias, float* __restrict__ C,
                                    int M, int N, int K, int relu) {
  const long total = (long)M * N;
  for (long e = blockIdx.x * (long)blockDim.x + threadIdx.x; e < total;
       e += (long)gridDim.x * blockDim.x) {
    const int m = (int)(e / N), n = (int)(e % N);
    float v = bias[n];
    for (int k = 0; k < K; ++k) v += A[(size_t)m*K + k] * W[(size_t)k*N + n];
    if (relu) v = fmaxf(v, 0.0f);
    C[e] = v;
  }
}

__global__ void concat2_kernel(const float* __restrict__ A, int Ca,
                               const float* __restrict__ Bp, int Cb,
                               float* __restrict__ C, long rows) {
  const int Cc = Ca + Cb;
  const long total = rows * Cc;
  for (long e = blockIdx.x * (long)blockDim.x + threadIdx.x; e < total;
       e += (long)gridDim.x * blockDim.x) {
    const long r = e / Cc; const int k = (int)(e % Cc);
    C[e] = (k < Ca) ? A[r * Ca + k] : Bp[r * Cb + (k - Ca)];
  }
}

__global__ void bcastcat_kernel(const float* __restrict__ gx, int Cg,
                                const float* __restrict__ X, int Cx,
                                float* __restrict__ C, int rowsPerB, long rows) {
  const int Cc = Cg + Cx;
  const long total = rows * Cc;
  for (long e = blockIdx.x * (long)blockDim.x + threadIdx.x; e < total;
       e += (long)gridDim.x * blockDim.x) {
    const long r = e / Cc; const int k = (int)(e % Cc);
    const int b = (int)(r / rowsPerB);
    C[e] = (k < Cg) ? gx[(size_t)b * Cg + k] : X[r * Cx + (k - Cg)];
  }
}

__global__ void maxpool_rows_kernel(const float* __restrict__ X, float* __restrict__ out,
                                    int rowsPer, int C, int Btot) {
  const long total = (long)Btot * C;
  for (long e = blockIdx.x * (long)blockDim.x + threadIdx.x; e < total;
       e += (long)gridDim.x * blockDim.x) {
    const int b = (int)(e / C), ch = (int)(e % C);
    float m = -3.4e38f;
    for (int r = 0; r < rowsPer; ++r) m = fmaxf(m, X[((size_t)b*rowsPer + r)*C + ch]);
    out[e] = m;
  }
}

__global__ void knn3_kernel(const float* __restrict__ pd, const float* __restrict__ ps,
                            int T, int S, int* __restrict__ idx3, float* __restrict__ w3) {
  const long total = (long)BB * T;
  for (long e = blockIdx.x * (long)blockDim.x + threadIdx.x; e < total;
       e += (long)gridDim.x * blockDim.x) {
    const int b = (int)(e / T);
    const float x = pd[e*3], y = pd[e*3+1], z = pd[e*3+2];
    const float* psb = ps + (size_t)b * S * 3;
    float d0 = 3.4e38f, d1 = 3.4e38f, d2v = 3.4e38f;
    int i0 = 0, i1 = 0, i2 = 0;
    for (int j = 0; j < S; ++j) {
      const float dx = psb[3*j]-x, dy = psb[3*j+1]-y, dz = psb[3*j+2]-z;
      const float d = dx*dx + dy*dy + dz*dz;
      if (d < d0)       { d2v = d1; i2 = i1; d1 = d0; i1 = i0; d0 = d; i0 = j; }
      else if (d < d1)  { d2v = d1; i2 = i1; d1 = d;  i1 = j; }
      else if (d < d2v) { d2v = d;  i2 = j; }
    }
    const float w0 = 1.0f/fmaxf(d0, 1e-16f), w1 = 1.0f/fmaxf(d1, 1e-16f),
                w2 = 1.0f/fmaxf(d2v, 1e-16f);
    const float s = w0 + w1 + w2;
    idx3[e*3] = i0; idx3[e*3+1] = i1; idx3[e*3+2] = i2;
    w3[e*3] = w0/s; w3[e*3+1] = w1/s; w3[e*3+2] = w2/s;
  }
}

__global__ void interp3_kernel(const float* __restrict__ xs, const int* __restrict__ idx3,
                               const float* __restrict__ w3, float* __restrict__ out,
                               int T, int S, int C) {
  const long total = (long)BB * T * C;
  for (long e = blockIdx.x * (long)blockDim.x + threadIdx.x; e < total;
       e += (long)gridDim.x * blockDim.x) {
    const long r = e / C; const int ch = (int)(e % C);
    const int b = (int)(r / T);
    float v = 0.0f;
#pragma unroll
    for (int j = 0; j < 3; ++j) {
      const int si = idx3[r*3 + j];
      v += w3[r*3 + j] * xs[((size_t)b*S + si)*C + ch];
    }
    out[e] = v;
  }
}

// ---------------------------------------------------------------------------
// Per-cloud log_softmax + argmax + BCE partial sum.
// ---------------------------------------------------------------------------
__global__ void __launch_bounds__(256)
softmax_loss_kernel(const float* __restrict__ a, const float* __restrict__ agt,
                    float* __restrict__ dist, int* __restrict__ apidx,
                    float* __restrict__ bce_partial) {
  const int b = blockIdx.x, t = threadIdx.x;
  const float* ab = a + (size_t)b * PP;
  __shared__ float rv[256];
  __shared__ int   ri[256];
  float bv = -3.4e38f; int bi = PP - 1;
  for (int j = t; j < PP; j += 256) {
    const float v = ab[j];
    if (v > bv) { bv = v; bi = j; }
  }
  rv[t] = bv; ri[t] = bi; __syncthreads();
  for (int s = 128; s > 0; s >>= 1) {
    if (t < s) {
      if (rv[t+s] > rv[t] || (rv[t+s] == rv[t] && ri[t+s] < ri[t])) {
        rv[t] = rv[t+s]; ri[t] = ri[t+s];
      }
    }
    __syncthreads();
  }
  const float mx = rv[0];
  if (t == 0) apidx[b] = ri[0];
  __syncthreads();
  float se = 0.0f;
  for (int j = t; j < PP; j += 256) se += __expf(ab[j] - mx);
  rv[t] = se; __syncthreads();
  for (int s = 128; s > 0; s >>= 1) { if (t < s) rv[t] += rv[t+s]; __syncthreads(); }
  const float lse = logf(rv[0]);
  __syncthreads();
  const float* gb = agt + (size_t)b * PP;
  float bsum = 0.0f;
  for (int j = t; j < PP; j += 256) {
    const float d = ab[j] - mx - lse;
    dist[(size_t)b*PP + j] = d;
    const float p  = 1.0f / (1.0f + __expf(-d));
    const float tt = rintf(gb[j]);
    bsum += -(tt * logf(p + 1e-12f) + (1.0f - tt) * logf(1.0f - p + 1e-12f));
  }
  rv[t] = bsum; __syncthreads();
  for (int s = 128; s > 0; s >>= 1) { if (t < s) rv[t] += rv[t+s]; __syncthreads(); }
  if (t == 0) bce_partial[b] = rv[0];
}

__global__ void gather_ap_kernel(const float* __restrict__ pos, const float* __restrict__ pg,
                                 const int* __restrict__ apidx,
                                 float* __restrict__ grasp_gt, float* __restrict__ ap_pts) {
  const int t = threadIdx.x;
  if (t < BB * 16) {
    const int b = t / 16, j = t % 16;
    grasp_gt[t] = pg[((size_t)b*PP + apidx[b])*16 + j];
  }
  if (t < BB * 3) {
    const int b = t / 3, j = t % 3;
    ap_pts[t] = pos[((size_t)b*PP + apidx[b])*3 + j];
  }
}

__global__ void __launch_bounds__(128)
final_loss_kernel(const float* __restrict__ gp, const float* __restrict__ gt,
                  const float* __restrict__ bce_partial, float* __restrict__ out2) {
  __shared__ float rv[128];
  const int t = threadIdx.x;
  const float d = gp[t] - gt[t];
  rv[t] = d * d;
  __syncthreads();
  for (int s = 64; s > 0; s >>= 1) { if (t < s) rv[t] += rv[t+s]; __syncthreads(); }
  if (t == 0) {
    out2[0] = rv[0] / 128.0f;
    float s = 0.0f;
    for (int b = 0; b < BB; ++b) s += bce_partial[b];
    out2[1] = s / (float)(BB * PP);
  }
}

// ---------------------------------------------------------------------------
extern "C" void kernel_launch(void* const* d_in, const int* in_sizes, int n_in,
                              void* d_out, int out_size, void* d_ws, size_t ws_size,
                              hipStream_t stream) {
  (void)in_sizes; (void)n_in; (void)out_size; (void)ws_size;
  auto fp = [&](int i) { return (const float*)d_in[i]; };
  const float* pos    = fp(0);
  const float* pgrasp = fp(1);
  const float* agt    = fp(2);
  const float *sa1W1=fp(3),  *sa1b1=fp(4),  *sa1W2=fp(5),  *sa1b2=fp(6),  *sa1W3=fp(7),  *sa1b3=fp(8);
  const float *sa2W1=fp(9),  *sa2b1=fp(10), *sa2W2=fp(11), *sa2b2=fp(12), *sa2W3=fp(13), *sa2b3=fp(14);
  const float *sa3W1=fp(15), *sa3b1=fp(16), *sa3W2=fp(17), *sa3b2=fp(18), *sa3W3=fp(19), *sa3b3=fp(20);
  const float *fp3W1=fp(21), *fp3b1=fp(22), *fp3W2=fp(23), *fp3b2=fp(24);
  const float *fp2W1=fp(25), *fp2b1=fp(26), *fp2W2=fp(27), *fp2b2=fp(28);
  const float *fp1W1=fp(29), *fp1b1=fp(30), *fp1W2=fp(31), *fp1b2=fp(32), *fp1W3=fp(33), *fp1b3=fp(34);
  const float *hdW1 =fp(35), *hdb1 =fp(36), *hdW2 =fp(37), *hdb2 =fp(38), *hdW3 =fp(39), *hdb3 =fp(40);
  const float *apW1 =fp(41), *apb1 =fp(42), *apW2 =fp(43), *apb2 =fp(44);
  const float *gpW1 =fp(45), *gpb1 =fp(46), *gpW2 =fp(47), *gpb2 =fp(48), *gpW3 =fp(49), *gpb3 =fp(50);
  float* out = (float*)d_out;

  char* wsb = (char*)d_ws;
  size_t off = 0;
  auto alloc = [&](size_t bytes) -> char* {
    char* p = wsb + off;
    off += (bytes + 255) & ~(size_t)255;
    return p;
  };
  auto cdiv = [](long a, long b) { return (int)((a + b - 1) / b); };
  auto gs   = [&](long total) { int b = cdiv(total, 256); return b > 32768 ? 32768 : b; };

  // ---- swizzled f16 weights (one prep kernel per layer, ~4 MB total)
  auto swz = [&](const float* W, int K, int N) -> _Float16* {
    const int KP = ((K + 31) / 32) * 32;
    _Float16* p = (_Float16*)alloc((size_t)KP * N * 2);
    swz_weight_kernel<<<gs((long)KP * N), 256, 0, stream>>>(W, p, K, N);
    return p;
  };
  _Float16 *s1w1 = swz(sa1W1, 6, 64),     *s1w2 = swz(sa1W2, 64, 64),   *s1w3 = swz(sa1W3, 64, 128);
  _Float16 *s2w1 = swz(sa2W1, 131, 128),  *s2w2 = swz(sa2W2, 128, 128), *s2w3 = swz(sa2W3, 128, 256);
  _Float16 *s3w1 = swz(sa3W1, 259, 256),  *s3w2 = swz(sa3W2, 256, 512), *s3w3 = swz(sa3W3, 512, 1024);
  _Float16 *f3w1 = swz(fp3W1, 1280, 256), *f3w2 = swz(fp3W2, 256, 256);
  _Float16 *f2w1 = swz(fp2W1, 384, 256),  *f2w2 = swz(fp2W2, 256, 128);
  _Float16 *f1w1 = swz(fp1W1, 131, 128),  *f1w2 = swz(fp1W2, 128, 128), *f1w3 = swz(fp1W3, 128, 128);
  _Float16 *hw1  = swz(hdW1, 128, 128),   *hw2  = swz(hdW2, 128, 128);
  _Float16 *gw1  = swz(gpW1, 1088, 512),  *gw2  = swz(gpW2, 512, 256);

  const long MS1 = (long)BB * SS1, MS2 = (long)BB * SS2, MP = (long)BB * PP;
  int*   idx1 = (int*)  alloc((size_t)MS1 * 4);
  float* pos1 = (float*)alloc((size_t)MS1 * 3 * 4);
  int*   idx2 = (int*)  alloc((size_t)MS2 * 4);
  float* pos2 = (float*)alloc((size_t)MS2 * 3 * 4);
  float* x1   = (float*)alloc((size_t)MS1 * 128 * 4);
  float* x2   = (float*)alloc((size_t)MS2 * 256 * 4);
  float* gx   = (float*)alloc((size_t)BB * GFD * 4);
  float* f3   = (float*)alloc((size_t)MS2 * 256 * 4);
  float* f2   = (float*)alloc((size_t)MS1 * 128 * 4);
  float* ascore=(float*)alloc((size_t)MP * 4);
  int*   nbr  = (int*)  alloc((size_t)MS1 * MAXNB * 4);
  int*   nval = (int*)  alloc((size_t)MS1 * 4);
  int*   kidx = (int*)  alloc((size_t)MP * 3 * 4);
  float* kw   = (float*)alloc((size_t)MP * 3 * 4);
  int*   apidx= (int*)  alloc(BB * 4);
  float* bcep = (float*)alloc(BB * 4);
  float* ggt  = (float*)alloc(BB * 16 * 4);
  float* appt = (float*)alloc(BB * 3 * 4);
  float* af1  = (float*)alloc(BB * 32 * 4);
  float* af   = (float*)alloc(BB * 64 * 4);
  float* gpin = (float*)alloc(BB * 1088 * 4);
  float* gph1 = (float*)alloc(BB * 512 * 4);
  float* gph2 = (float*)alloc(BB * 256 * 4);
  const size_t AR = (size_t)MP * 131 * 4;   // 34.3 MB ping-pong arenas
  float* T0 = (float*)alloc(AR);
  float* T1 = (float*)alloc(AR);

  // ---- SA1: FPS P->S1, radius group, fused WMMA MLP [6->64->64->128] + max
  fps_kernel<<<BB, 256, 0, stream>>>(pos, PP, SS1, idx1, pos1);
  neighbor_kernel<<<(int)MS1, 256, 0, stream>>>(pos1, pos, PP, SS1, 0.2f*0.2f, nbr, nval);
  sa_mlp_kernel<3, 32, 64, 64, 128><<<(int)MS1, 128, 0, stream>>>(
      pos, pos, PP, SS1, pos1, nbr, nval,
      s1w1, sa1b1, s1w2, sa1b2, s1w3, sa1b3, x1);

  // ---- SA2: FPS S1->S2, fused WMMA MLP [131->128->128->256] + max
  fps_kernel<<<BB, 256, 0, stream>>>(pos1, SS1, SS2, idx2, pos2);
  neighbor_kernel<<<(int)MS2, 256, 0, stream>>>(pos2, pos1, SS1, SS2, 0.4f*0.4f, nbr, nval);
  sa_mlp_kernel<128, 160, 128, 128, 256><<<(int)MS2, 128, 0, stream>>>(
      x1, pos1, SS1, SS2, pos2, nbr, nval,
      s2w1, sa2b1, s2w2, sa2b2, s2w3, sa2b3, x2);

  // ---- SA3 global MLP [259->256->512->1024] + global max pool
  concat2_kernel<<<gs(MS2*259), 256, 0, stream>>>(x2, 256, pos2, 3, T0, MS2);
  gemm_sw_kernel<<<dim3(4, cdiv(MS2,64)), 128, 0, stream>>>(T0, s3w1, sa3b1, T1, (int)MS2, 256, 259, 1);
  gemm_sw_kernel<<<dim3(8, cdiv(MS2,64)), 128, 0, stream>>>(T1, s3w2, sa3b2, T0, (int)MS2, 512, 256, 1);
  gemm_sw_kernel<<<dim3(16, cdiv(MS2,64)), 128, 0, stream>>>(T0, s3w3, sa3b3, T1, (int)MS2, GFD, 512, 0);
  maxpool_rows_kernel<<<gs((long)BB*GFD), 256, 0, stream>>>(T1, gx, SS2, GFD, BB);

  // ---- FP3: broadcast gx, MLP [1280->256->256]
  bcastcat_kernel<<<gs(MS2*1280), 256, 0, stream>>>(gx, GFD, x2, 256, T0, SS2, MS2);
  gemm_sw_kernel<<<dim3(4, cdiv(MS2,64)), 128, 0, stream>>>(T0, f3w1, fp3b1, T1, (int)MS2, 256, 1280, 1);
  gemm_sw_kernel<<<dim3(4, cdiv(MS2,64)), 128, 0, stream>>>(T1, f3w2, fp3b2, f3, (int)MS2, 256, 256, 0);

  // ---- FP2: 3NN interp S2->S1, MLP [384->256->128]
  knn3_kernel<<<gs(MS1), 256, 0, stream>>>(pos1, pos2, SS1, SS2, kidx, kw);
  interp3_kernel<<<gs(MS1*256), 256, 0, stream>>>(f3, kidx, kw, T0, SS1, SS2, 256);
  concat2_kernel<<<gs(MS1*384), 256, 0, stream>>>(T0, 256, x1, 128, T1, MS1);
  gemm_sw_kernel<<<dim3(4, cdiv(MS1,64)), 128, 0, stream>>>(T1, f2w1, fp2b1, T0, (int)MS1, 256, 384, 1);
  gemm_sw_kernel<<<dim3(2, cdiv(MS1,64)), 128, 0, stream>>>(T0, f2w2, fp2b2, f2, (int)MS1, 128, 256, 0);

  // ---- FP1: 3NN interp S1->P, MLP [131->128->128->128]
  knn3_kernel<<<gs(MP), 256, 0, stream>>>(pos, pos1, PP, SS1, kidx, kw);
  interp3_kernel<<<gs(MP*128), 256, 0, stream>>>(f2, kidx, kw, T0, PP, SS1, 128);
  concat2_kernel<<<gs(MP*131), 256, 0, stream>>>(T0, 128, pos, 3, T1, MP);
  gemm_sw_kernel<<<dim3(2, cdiv(MP,64)), 128, 0, stream>>>(T1, f1w1, fp1b1, T0, (int)MP, 128, 131, 1);
  gemm_sw_kernel<<<dim3(2, cdiv(MP,64)), 128, 0, stream>>>(T0, f1w2, fp1b2, T1, (int)MP, 128, 128, 1);
  gemm_sw_kernel<<<dim3(2, cdiv(MP,64)), 128, 0, stream>>>(T1, f1w3, fp1b3, T0, (int)MP, 128, 128, 0);

  // ---- head [128->128->128->1], log_softmax + argmax + BCE
  gemm_sw_kernel<<<dim3(2, cdiv(MP,64)), 128, 0, stream>>>(T0, hw1, hdb1, T1, (int)MP, 128, 128, 1);
  gemm_sw_kernel<<<dim3(2, cdiv(MP,64)), 128, 0, stream>>>(T1, hw2, hdb2, T0, (int)MP, 128, 128, 1);
  linear_small_kernel<<<gs(MP), 256, 0, stream>>>(T0, hdW3, hdb3, ascore, (int)MP, 1, 128, 0);
  softmax_loss_kernel<<<BB, 256, 0, stream>>>(ascore, agt, out + 128, apidx, bcep);
  gather_ap_kernel<<<1, 256, 0, stream>>>(pos, pgrasp, apidx, ggt, appt);

  // ---- approach encoder [3->32->64] and grasp predictor [1088->512->256->16]
  linear_small_kernel<<<1, 256, 0, stream>>>(appt, apW1, apb1, af1, BB, 32, 3, 1);
  linear_small_kernel<<<2, 256, 0, stream>>>(af1, apW2, apb2, af, BB, 64, 32, 0);
  concat2_kernel<<<gs((long)BB*1088), 256, 0, stream>>>(gx, 1024, af, 64, gpin, BB);
  gemm_sw_kernel<<<dim3(8, 1), 128, 0, stream>>>(gpin, gw1, gpb1, gph1, BB, 512, 1088, 1);
  gemm_sw_kernel<<<dim3(4, 1), 128, 0, stream>>>(gph1, gw2, gpb2, gph2, BB, 256, 512, 1);
  linear_small_kernel<<<1, 256, 0, stream>>>(gph2, gpW3, gpb3, out, BB, 16, 256, 0);

  // ---- scalar losses
  final_loss_kernel<<<1, 128, 0, stream>>>(out, ggt, bcep, out + 128 + (size_t)MP);
}
